// PanguModel_20590073217477
// MI455X (gfx1250) — matrix operand order repaired and checked by reading, not compiled, not run on hardware
//
#include <hip/hip_runtime.h>
#include <hip/hip_bf16.h>
#include <math.h>

// ---------------------------------------------------------------------------
// Pangu EarthSpecificBlock on MI455X (gfx1250, wave32, WMMA).
// f16 operands + f32 WMMA accumulate (v_wmma_f32_16x16x32_f16).
// Tile staging uses CDNA5 async global->LDS (ASYNCcnt) with double buffering.
// Attention fully fused per (window, head); scores live in LDS (~118KB/WGP).
// ---------------------------------------------------------------------------

typedef __attribute__((ext_vector_type(16))) _Float16 v16h;
typedef __attribute__((ext_vector_type(8)))  float    v8f;
typedef __attribute__((ext_vector_type(4)))  float    f32x4;
typedef __attribute__((ext_vector_type(4)))  _Float16 f16x4;

#define DIM_Z 8
#define DIM_H 96
#define DIM_W 180
#define CH    192
#define NTOK  144          // tokens per window (2*6*12)
#define NWIN  960          // 4*16*15 windows
#define NHEAD 6
#define LTOK  138240       // Z*H*W

static __device__ inline v8f wmma_f16(v16h a, v16h b, v8f c) {
  return __builtin_amdgcn_wmma_f32_16x16x32_f16(false, a, false, b, (short)0, c,
                                                false, false);
}

// Low 32 bits of a generic LDS pointer are the LDS byte offset (aperture map).
static __device__ inline unsigned lds_lo(const void* p) {
  return (unsigned)(size_t)p;
}

// CDNA5 async copy: global -> LDS, 16 bytes, tracked by ASYNCcnt.
static __device__ inline void async_ld_b128(unsigned lds_off, const void* gaddr) {
  asm volatile("global_load_async_to_lds_b128 %0, %1, off"
               :: "v"(lds_off), "v"((unsigned long long)(size_t)gaddr)
               : "memory");
}
static __device__ inline void wait_async0() {
  asm volatile("s_wait_asynccnt 0x0" ::: "memory");
}

// A-matrix 16x32 f16 fragment (ISA 7.12.2): lane half h holds K chunks
// [8h..8h+7] and [8h+16..8h+23] of row (lane&15).
static __device__ inline v16h load_fragA(const _Float16* base, int stride, int lane) {
  int row = lane & 15, half = lane >> 4;
  const _Float16* p = base + row * stride + half * 8;
  union { v16h v; f32x4 f[2]; } u;
  u.f[0] = *(const f32x4*)(p);
  u.f[1] = *(const f32x4*)(p + 16);
  return u.v;
}

// B-matrix 32x16 f16 fragment: lane = column, lane half h holds K chunks
// [8h..8h+7] and [8h+16..8h+23]. `base` points at a row-major [col][k] (i.e. transposed) tile.
static __device__ inline v16h load_fragB(const _Float16* base, int stride, int lane) {
  int col = lane & 15, half = lane >> 4;
  const _Float16* p = base + col * stride + half * 8;
  union { v16h v; f32x4 f[2]; } u;
  u.f[0] = *(const f32x4*)(p);
  u.f[1] = *(const f32x4*)(p + 16);
  return u.v;
}

// ---------------------------------------------------------------------------
// Weight prep: f32 (K,Nc) -> f16 transposed (Nc,K) so GEMM B-tiles are
// contiguous [col][k] rows stageable with 128-bit async copies.
// ---------------------------------------------------------------------------
__global__ __launch_bounds__(256) void cvt_transpose(const float* __restrict__ src,
                                                     _Float16* __restrict__ dst,
                                                     int K, int Nc) {
  int i = blockIdx.x * 256 + threadIdx.x;
  if (i >= K * Nc) return;
  int k = i / Nc, n = i % Nc;
  dst[(size_t)n * K + k] = (_Float16)src[i];
}

// ---------------------------------------------------------------------------
// Window partition (with optional roll) : x f32 (Z,H,W,C) -> xw f16 (win,n,C)
// ---------------------------------------------------------------------------
__global__ __launch_bounds__(256) void partition_kernel(const float* __restrict__ x,
                                                        _Float16* __restrict__ xw,
                                                        const int* __restrict__ rollp) {
  int g = blockIdx.x * 256 + threadIdx.x;          // one group = 4 channels
  const int TOT = LTOK * (CH / 4);
  if (g >= TOT) return;
  int d = g / (CH / 4);
  int c = (g % (CH / 4)) * 4;
  int win = d / NTOK, n = d % NTOK;
  int zb = win / 240, hb = (win / 15) % 16, wb = win % 15;
  int z1 = n / 72, h1 = (n / 12) % 6, w1 = n % 12;
  int z = zb * 2 + z1, hh = hb * 6 + h1, w = wb * 12 + w1;
  if (*rollp) {                                    // xr[i] = x[(i - shift) % dim]
    z = (z + DIM_Z - 1) % DIM_Z;
    hh = (hh + DIM_H - 3) % DIM_H;
    w = (w + DIM_W - 6) % DIM_W;
  }
  size_t src = ((size_t)(z * DIM_H + hh) * DIM_W + w) * CH + c;
  f32x4 v = *(const f32x4*)(x + src);
  f16x4 o;
  o.x = (_Float16)v.x; o.y = (_Float16)v.y; o.z = (_Float16)v.z; o.w = (_Float16)v.w;
  *(f16x4*)(xw + (size_t)d * CH + c) = o;
}

// ---------------------------------------------------------------------------
// f16 GEMM: out[M,Nc] = A[M,K] @ Wt[Nc,K]^T + bias, WMMA f32 accumulate.
// mode 0: f16 out; mode 1: f16 out + exact GELU; mode 2: f32 out.
// Block = 256 threads (8 waves), tile BM=128 x BN=64, BK=32.
// Double-buffered async global->LDS staging (ASYNCcnt).
// ---------------------------------------------------------------------------
#define BM 128
#define BN 64
#define BK 32
__global__ __launch_bounds__(256) void gemm_f16(const _Float16* __restrict__ A,
                                                const _Float16* __restrict__ Wt,
                                                const float* __restrict__ bias,
                                                void* __restrict__ out,
                                                int M, int Nc, int K, int mode) {
  __shared__ _Float16 As[2][BM * BK];   // 2 x 8 KB, row-major [m][k]
  __shared__ _Float16 Ws[2][BN * BK];   // 2 x 4 KB, row-major [n][k]
  int tid = threadIdx.x, lane = tid & 31, wid = tid >> 5;
  int m0 = blockIdx.y * BM, n0 = blockIdx.x * BN;
  int wm = (wid >> 1) * 32, wn = (wid & 1) * 32;

  v8f acc[2][2];
  for (int i = 0; i < 2; i++)
    for (int j = 0; j < 2; j++)
      acc[i][j] = (v8f){0.f, 0.f, 0.f, 0.f, 0.f, 0.f, 0.f, 0.f};

  // Async-issue one K-tile (A: 512 chunks, W: 256 chunks of 16B) into buf b.
  auto issue_tile = [&](int k0, int b) {
    for (int c = tid; c < (BM * BK / 8); c += 256) {
      int row = c >> 2, seg = (c & 3) * 8;
      async_ld_b128(lds_lo(&As[b][row * BK + seg]),
                    A + (size_t)(m0 + row) * K + k0 + seg);
    }
    {
      int row = tid >> 2, seg = (tid & 3) * 8;    // 256 chunks: 64 rows x 4 segs
      async_ld_b128(lds_lo(&Ws[b][row * BK + seg]),
                    Wt + (size_t)(n0 + row) * K + k0 + seg);
    }
  };

  int nk = K / BK;
  issue_tile(0, 0);
  for (int t = 0; t < nk; t++) {
    wait_async0();            // drain this wave's async copies
    __syncthreads();          // publish all waves' LDS writes
    if (t + 1 < nk) issue_tile((t + 1) * BK, (t + 1) & 1);
    const _Float16* Ab = As[t & 1];
    const _Float16* Wb = Ws[t & 1];
    v16h a0 = load_fragA(Ab + (wm + 0) * BK, BK, lane);
    v16h a1 = load_fragA(Ab + (wm + 16) * BK, BK, lane);
    v16h b0 = load_fragB(Wb + (wn + 0) * BK, BK, lane);
    v16h b1 = load_fragB(Wb + (wn + 16) * BK, BK, lane);
    acc[0][0] = wmma_f16(a0, b0, acc[0][0]);
    acc[0][1] = wmma_f16(a0, b1, acc[0][1]);
    acc[1][0] = wmma_f16(a1, b0, acc[1][0]);
    acc[1][1] = wmma_f16(a1, b1, acc[1][1]);
  }

  int col = lane & 15, half = lane >> 4;
  for (int i = 0; i < 2; i++)
    for (int j = 0; j < 2; j++) {
      int gn = n0 + wn + j * 16 + col;
      float bv = bias[gn];
      for (int r = 0; r < 8; r++) {
        int gm = m0 + wm + i * 16 + half * 8 + r;
        float v = acc[i][j][r] + bv;
        if (mode == 1) v = 0.5f * v * (1.f + erff(v * 0.70710678f));
        if (mode == 2)
          ((float*)out)[(size_t)gm * Nc + gn] = v;
        else
          ((_Float16*)out)[(size_t)gm * Nc + gn] = (_Float16)v;
      }
    }
}

// ---------------------------------------------------------------------------
// Fused attention: one block per (window, head). Q,K,V + 144x160 score
// matrix stay in LDS (120832 B). S = scale*(Q K^T) + earth-bias, softmax,
// O = P V, all via WMMA 16x16x32. Q/K staged with async copies.
// ---------------------------------------------------------------------------
__global__ __launch_bounds__(256) void attn_kernel(const _Float16* __restrict__ qkv,
                                                   const float* __restrict__ bias_table,
                                                   _Float16* __restrict__ attnout) {
  extern __shared__ char smem[];
  _Float16* Qs = (_Float16*)smem;        // 144 x 32
  _Float16* Ks = Qs + NTOK * 32;         // 144 x 32
  _Float16* Vt = Ks + NTOK * 32;         // 32 x 160 (transposed, padded)
  float*    S  = (float*)(Vt + 32 * 160);// 144 x 160 (K-padded with zeros)

  int win = blockIdx.x / NHEAD, head = blockIdx.x % NHEAD;
  int tid = threadIdx.x, lane = tid & 31, wid = tid >> 5;
  int tow = win / 15;                    // windows ordered (z,h,wlon)
  const _Float16* base = qkv + (size_t)win * NTOK * (3 * CH) + head * 32;

  // ---- async-stage Q and K rows (144 x 32 halves each, 16B chunks) ----
  for (int c = tid; c < NTOK * 4; c += 256) {
    int n = c >> 2, seg = (c & 3) * 8;
    async_ld_b128(lds_lo(Qs + n * 32 + seg), base + (size_t)n * (3 * CH) + seg);
    async_ld_b128(lds_lo(Ks + n * 32 + seg), base + (size_t)n * (3 * CH) + CH + seg);
  }
  // ---- load V transposed [d][m], zero-pad m=144..159 ----
  for (int i = tid; i < 32 * 160; i += 256) {
    int dd = i / 160, m = i % 160;
    Vt[dd * 160 + m] =
        (m < NTOK) ? base[(size_t)m * (3 * CH) + 2 * CH + dd] : (_Float16)0.f;
  }
  wait_async0();
  __syncthreads();

  // ---- stage 1: S = scale*(Q K^T) + bias ; 81 tiles, K=32 in one WMMA ----
  for (int t = wid; t < 81; t += 8) {
    int ti = t / 9, tj = t % 9;
    v16h a = load_fragA(Qs + ti * 16 * 32, 32, lane);
    v16h b = load_fragB(Ks + tj * 16 * 32, 32, lane);
    v8f z = (v8f){0.f, 0.f, 0.f, 0.f, 0.f, 0.f, 0.f, 0.f};
    v8f s = wmma_f16(a, b, z);
    int mk = tj * 16 + (lane & 15);
    int z2 = mk / 72, h2 = (mk / 12) % 6, w2 = mk % 12;
    for (int r = 0; r < 8; r++) {
      int nq = ti * 16 + (lane >> 4) * 8 + r;
      int z1 = nq / 72, h1 = (nq / 12) % 6, w1 = nq % 12;
      int idx = 828 * (z1 + 2 * z2) + 23 * (h1 + 6 * h2) + (w1 - w2 + 11);
      S[nq * 160 + mk] = s[r] * 0.17677669f +
          bias_table[(size_t)idx * (64 * NHEAD) + tow * NHEAD + head];
    }
  }
  __syncthreads();

  // ---- softmax over rows (wave32 shfl reductions), zero the K-pad ----
  for (int rr = 0; rr < 18; rr++) {
    int row = wid * 18 + rr;
    float* srow = S + row * 160;
    float vals[5];
    float mx = -1e30f;
    for (int k = 0; k < 5; k++) {
      int c = lane + 32 * k;
      vals[k] = (c < NTOK) ? srow[c] : -1e30f;
      mx = fmaxf(mx, vals[k]);
    }
    for (int o = 16; o > 0; o >>= 1) mx = fmaxf(mx, __shfl_xor(mx, o));
    float sum = 0.f;
    for (int k = 0; k < 5; k++) {
      float e = (lane + 32 * k < NTOK) ? __expf(vals[k] - mx) : 0.f;
      vals[k] = e;
      sum += e;
    }
    for (int o = 16; o > 0; o >>= 1) sum += __shfl_xor(sum, o);
    float inv = 1.f / sum;
    for (int k = 0; k < 5; k++) {
      int c = lane + 32 * k;
      if (c < NTOK) srow[c] = vals[k] * inv;
    }
    if (lane < 16) srow[NTOK + lane] = 0.f;
  }
  __syncthreads();

  // ---- stage 2: O = P V ; 9x2 tiles, K loop over 160 (padded) ----
  for (int t = wid; t < 18; t += 8) {
    int ti = t >> 1, tj = t & 1;
    v8f acc = (v8f){0.f, 0.f, 0.f, 0.f, 0.f, 0.f, 0.f, 0.f};
    for (int ks = 0; ks < 5; ks++) {
      union { v16h v; _Float16 h[16]; } ua;
      int row = lane & 15, half = lane >> 4;
      const float* sr = S + (ti * 16 + row) * 160 + ks * 32 + half * 8;
      for (int e = 0; e < 8; e++) {
        ua.h[e]     = (_Float16)sr[e];
        ua.h[e + 8] = (_Float16)sr[e + 16];
      }
      v16h b = load_fragB(Vt + (tj * 16) * 160 + ks * 32, 160, lane);
      acc = wmma_f16(ua.v, b, acc);
    }
    int nd = tj * 16 + (lane & 15), half = lane >> 4;
    for (int r = 0; r < 8; r++) {
      int nq = ti * 16 + half * 8 + r;
      attnout[((size_t)win * NTOK + nq) * CH + head * 32 + nd] = (_Float16)acc[r];
    }
  }
}

// ---------------------------------------------------------------------------
// Window reverse + LayerNorm + residual: d_out = x + LN(reverse(proj))
// Also emits x1 as f16 for the MLP GEMM. One wave per token.
// ---------------------------------------------------------------------------
__global__ __launch_bounds__(256) void reverse_ln_kernel(const float* __restrict__ x,
                                                         const float* __restrict__ proj,
                                                         const float* __restrict__ gamma,
                                                         const float* __restrict__ beta,
                                                         const int* __restrict__ rollp,
                                                         float* __restrict__ xout,
                                                         _Float16* __restrict__ xout_h) {
  int wid = threadIdx.x >> 5, lane = threadIdx.x & 31;
  int t = blockIdx.x * 8 + wid;
  int z = t / (DIM_H * DIM_W), hh = (t / DIM_W) % DIM_H, w = t % DIM_W;
  if (*rollp) {  // inverse roll: result[i] = xr2[(i + shift) % dim]
    z = (z + 1) % DIM_Z;
    hh = (hh + 3) % DIM_H;
    w = (w + 6) % DIM_W;
  }
  int zb = z >> 1, z1 = z & 1, hb = hh / 6, h1 = hh % 6, wb = w / 12, w1 = w % 12;
  int win = (zb * 16 + hb) * 15 + wb;
  int n = (z1 * 6 + h1) * 12 + w1;
  const float* y = proj + ((size_t)win * NTOK + n) * CH;

  float v[6], s = 0.f, sq = 0.f;
  for (int k = 0; k < 6; k++) {
    float a = y[lane + 32 * k];
    v[k] = a; s += a; sq += a * a;
  }
  for (int o = 16; o > 0; o >>= 1) { s += __shfl_xor(s, o); sq += __shfl_xor(sq, o); }
  float mu = s * (1.f / CH);
  float rs = rsqrtf(sq * (1.f / CH) - mu * mu + 1e-5f);
  for (int k = 0; k < 6; k++) {
    int c = lane + 32 * k;
    float o = x[(size_t)t * CH + c] + (v[k] - mu) * rs * gamma[c] + beta[c];
    xout[(size_t)t * CH + c] = o;
    xout_h[(size_t)t * CH + c] = (_Float16)o;
  }
}

// ---------------------------------------------------------------------------
// Final: d_out += LN(h2) with gamma2/beta2. One wave per token.
// ---------------------------------------------------------------------------
__global__ __launch_bounds__(256) void final_ln_kernel(const float* __restrict__ h2,
                                                       const float* __restrict__ gamma,
                                                       const float* __restrict__ beta,
                                                       float* __restrict__ out) {
  int wid = threadIdx.x >> 5, lane = threadIdx.x & 31;
  int t = blockIdx.x * 8 + wid;
  const float* y = h2 + (size_t)t * CH;
  float v[6], s = 0.f, sq = 0.f;
  for (int k = 0; k < 6; k++) {
    float a = y[lane + 32 * k];
    v[k] = a; s += a; sq += a * a;
  }
  for (int o = 16; o > 0; o >>= 1) { s += __shfl_xor(s, o); sq += __shfl_xor(sq, o); }
  float mu = s * (1.f / CH);
  float rs = rsqrtf(sq * (1.f / CH) - mu * mu + 1e-5f);
  for (int k = 0; k < 6; k++) {
    int c = lane + 32 * k;
    out[(size_t)t * CH + c] += (v[k] - mu) * rs * gamma[c] + beta[c];
  }
}

// ---------------------------------------------------------------------------
extern "C" void kernel_launch(void* const* d_in, const int* in_sizes, int n_in,
                              void* d_out, int out_size, void* d_ws, size_t ws_size,
                              hipStream_t stream) {
  const float* x          = (const float*)d_in[0];
  const float* w_qkv      = (const float*)d_in[1];
  const float* b_qkv      = (const float*)d_in[2];
  const float* w_proj     = (const float*)d_in[3];
  const float* b_proj     = (const float*)d_in[4];
  const float* bias_table = (const float*)d_in[5];
  const float* gamma1     = (const float*)d_in[6];
  const float* beta1      = (const float*)d_in[7];
  const float* gamma2     = (const float*)d_in[8];
  const float* beta2      = (const float*)d_in[9];
  const float* w_mlp1     = (const float*)d_in[10];
  const float* b_mlp1     = (const float*)d_in[11];
  const float* w_mlp2     = (const float*)d_in[12];
  const float* b_mlp2     = (const float*)d_in[13];
  const int*   rollp      = (const int*)d_in[17];

  char* ws = (char*)d_ws;
  const size_t MB = 1ull << 20;
  _Float16* wqkv_t  = (_Float16*)(ws + 0);            // 216 KB  (576 x 192)
  _Float16* wproj_t = (_Float16*)(ws + 256 * 1024);   // 72 KB   (192 x 192)
  _Float16* wm1_t   = (_Float16*)(ws + 512 * 1024);   // 288 KB  (768 x 192)
  _Float16* wm2_t   = (_Float16*)(ws + 1 * MB);       // 288 KB  (192 x 768)
  _Float16* xw_h    = (_Float16*)(ws + 2 * MB);       // 53.1 MB ; reused as x1_f16
  _Float16* qkv_h   = (_Float16*)(ws + 64 * MB);      // 159.3 MB
  _Float16* attn_h  = (_Float16*)(ws + 224 * MB);     // 53.1 MB
  float*    proj_f  = (float*)   (ws + 280 * MB);     // 106.2 MB ; reused as h2
  _Float16* hmlp_h  = (_Float16*)(ws + 64 * MB);      // 212.3 MB (reuses qkv+attn)

  // 1. weights -> f16, transposed to (Nc, K)
  cvt_transpose<<<(110592 + 255) / 256, 256, 0, stream>>>(w_qkv, wqkv_t, CH, 576);
  cvt_transpose<<<(36864 + 255) / 256, 256, 0, stream>>>(w_proj, wproj_t, CH, CH);
  cvt_transpose<<<(147456 + 255) / 256, 256, 0, stream>>>(w_mlp1, wm1_t, CH, 768);
  cvt_transpose<<<(147456 + 255) / 256, 256, 0, stream>>>(w_mlp2, wm2_t, 768, CH);

  // 2. window partition
  partition_kernel<<<(LTOK * (CH / 4) + 255) / 256, 256, 0, stream>>>(x, xw_h, rollp);

  // 3. QKV GEMM: (138240 x 192) @ (192 x 576) -> f16
  gemm_f16<<<dim3(576 / BN, LTOK / BM), 256, 0, stream>>>(
      xw_h, wqkv_t, b_qkv, qkv_h, LTOK, 576, CH, 0);

  // 4. fused attention: 960 windows x 6 heads
  size_t attn_lds = (size_t)NTOK * 32 * 2 * 2 + 32 * 160 * 2 + (size_t)NTOK * 160 * 4;
  attn_kernel<<<NWIN * NHEAD, 256, attn_lds, stream>>>(qkv_h, bias_table, attn_h);

  // 5. proj GEMM -> f32
  gemm_f16<<<dim3(CH / BN, LTOK / BM), 256, 0, stream>>>(
      attn_h, wproj_t, b_proj, proj_f, LTOK, CH, CH, 2);

  // 6. window reverse + LN + residual -> d_out (f32) and x1 f16
  reverse_ln_kernel<<<LTOK / 8, 256, 0, stream>>>(x, proj_f, gamma1, beta1, rollp,
                                                  (float*)d_out, xw_h);

  // 7. MLP1 GEMM + GELU -> f16 hidden
  gemm_f16<<<dim3(768 / BN, LTOK / BM), 256, 0, stream>>>(
      xw_h, wm1_t, b_mlp1, hmlp_h, LTOK, 768, CH, 1);

  // 8. MLP2 GEMM -> f32 (reuses proj buffer)
  gemm_f16<<<dim3(CH / BN, LTOK / BM), 256, 0, stream>>>(
      hmlp_h, wm2_t, b_mlp2, proj_f, LTOK, CH, 768, 2);

  // 9. d_out += LN(h2)
  final_ln_kernel<<<LTOK / 8, 256, 0, stream>>>(proj_f, gamma2, beta2, (float*)d_out);
}
